// LSTMOpen_29222957482738
// MI455X (gfx1250) — matrix operand order repaired and checked by reading, not compiled, not run on hardware
//
#include <hip/hip_runtime.h>
#include <hip/hip_bf16.h>

typedef __bf16 bf16_t;
typedef __attribute__((ext_vector_type(16))) __bf16 v16bf;
typedef __attribute__((ext_vector_type(8)))  __bf16 v8bf;
typedef __attribute__((ext_vector_type(8)))  float  v8f;
typedef __attribute__((ext_vector_type(4)))  unsigned int u32x4;
typedef __attribute__((ext_vector_type(8)))  int i32x8;
typedef __attribute__((ext_vector_type(4)))  int i32x4;

// Problem constants
#define BB   64
#define SS   512
#define DD   400
#define DP   416          // D padded to multiple of 32 for bf16 WMMA K-loop
#define HH   1152         // multiple of 32
#define G4   4608         // 4*H
#define MM   32768        // B*S

// LDS layout in lstm_steps
#define WSLAB_BYTES   (4 * 16 * HH * 2)        // 147456: 4 gates x 16 rows x 1152 K bf16
#define XGTILE_BYTES  (4 * BB * 16 * 2)        // 8192: [gate][b][16 cols] bf16
#define LDS_TOTAL     (WSLAB_BYTES + 2 * XGTILE_BYTES)   // 163840

__device__ __forceinline__ float sigf(float x) {
    return 1.0f / (1.0f + __expf(-x));
}

// Build a 16-wide bf16 A-fragment from two contiguous 8-element chunks
// in contiguous K order (k = kb + 16*half + s), matching the B fragment.
__device__ __forceinline__ v16bf make_afrag(const bf16_t* __restrict__ rowptr,
                                            int kb, bool hiHalf) {
    int k0 = kb + (hiHalf ? 16 : 0);       // first  8 K values for this lane
    int k1 = k0 + 8;                       // second 8 K values for this lane
    v8bf lo = *(const v8bf*)(rowptr + k0);
    v8bf hi = *(const v8bf*)(rowptr + k1);
    return __builtin_shufflevector(lo, hi, 0,1,2,3,4,5,6,7,8,9,10,11,12,13,14,15);
}

// ---------------------------------------------------------------------------
// Tensor Data Mover: DMA one timestep's gate-preactivation tile
//   xg_t tile: z = 4 gates (stride HH elems), y = 64 batch rows (stride G4
//   elems), x = 16 bf16 cols (contiguous, 32B lines) -> LDS contiguous
//   [gate][b][x].  D# built per CDNA5 ISA.
// ---------------------------------------------------------------------------
__device__ __forceinline__ void tdm_load_xg_tile(const bf16_t* gaddr, unsigned lds_off) {
    unsigned long long ga = (unsigned long long)(uintptr_t)gaddr;
    u32x4 g0;
    g0[0] = 1u;                                   // count=1, user descriptor
    g0[1] = lds_off;                              // lds_addr (bytes)
    g0[2] = (unsigned)(ga & 0xFFFFFFFFu);         // global_addr[31:0]
    g0[3] = (unsigned)((ga >> 32) & 0x01FFFFFFu)  // global_addr[56:32]
          | (2u << 30);                           // type = 2 ("image")
    i32x8 g1;
    g1[0] = 0x00010000;                 // workgroup_mask=0, data_size=1 (2B)
    g1[1] = (int)(16u << 16);           // tensor_dim0 = 16  (bits 79:48)
    g1[2] = (int)(64u << 16);           // tensor_dim1 = 64  (bits 111:80)
    g1[3] = (int)(16u << 16);           // tile_dim0  = 16  (bits 127:112)
    g1[4] = (int)(64u | (4u << 16));    // tile_dim1 = 64, tile_dim2 = 4
    g1[5] = (int)G4;                    // tensor_dim0_stride[31:0] = 4608
    g1[6] = (int)((unsigned)HH << 16);  // tensor_dim1_stride[15:0] = 1152
    g1[7] = 0;                          // tensor_dim1_stride[47:16] = 0
    i32x4 g2 = {4, 0, 0, 0};            // tensor_dim2 = 4; no dim3
    i32x4 g3 = {0, 0, 0, 0};
#if __has_include(<hip/amd_detail/amd_gfx1250_TDM.h>)
    i32x8 g4 = {0, 0, 0, 0, 0, 0, 0, 0};
    __builtin_amdgcn_tensor_load_to_lds(g0, g1, g2, g3, g4, 0);
#else
    __builtin_amdgcn_tensor_load_to_lds(g0, g1, g2, g3, 0);
#endif
}

// ---------------------------------------------------------------------------
// Conversion / staging kernels
// ---------------------------------------------------------------------------
__global__ void cvt_pad_bf16(const float* __restrict__ src, bf16_t* __restrict__ dst,
                             int rows, int scols, int dcols) {
    long i = (long)blockIdx.x * blockDim.x + threadIdx.x;
    long n = (long)rows * dcols;
    if (i >= n) return;
    int r = (int)(i / dcols), c = (int)(i % dcols);
    float v = (c < scols) ? src[(long)r * scols + c] : 0.0f;
    dst[i] = (bf16_t)v;
}

__global__ void init_state(const float* __restrict__ h0, const float* __restrict__ c0,
                           bf16_t* __restrict__ hbuf0, float* __restrict__ cbuf,
                           int* sync_cnt) {
    int i = blockIdx.x * blockDim.x + threadIdx.x;
    if (i < BB * HH) {
        hbuf0[i] = (bf16_t)h0[i];
        cbuf[i]  = c0[i];
    }
    if (i == 0) *sync_cnt = 0;
}

// ---------------------------------------------------------------------------
// Phase 1: xg[s,b,n] = sum_d inp[b,s,d]*w_ih[n,d] + b_ih[n] + b_hh[n]  (bf16)
// [S,B,4H] layout -> each timestep's slab is contiguous for the scan phase.
// One wave per (16-row m-tile) x (4 adjacent 16-col n-tiles), K=416.
// ---------------------------------------------------------------------------
__global__ void __launch_bounds__(256) gemm_xg(
    const bf16_t* __restrict__ A,     // [MM, DP] bf16 (inp, padded)
    const bf16_t* __restrict__ Bw,    // [G4, DP] bf16 (w_ih, padded)
    const float* __restrict__ b_ih, const float* __restrict__ b_hh,
    bf16_t* __restrict__ xg)          // [SS, BB, G4] bf16
{
    int wid  = blockIdx.x * (blockDim.x >> 5) + (threadIdx.x >> 5);
    int lane = threadIdx.x & 31;
    int mt = wid / (G4 / 64);          // 0..2047
    int ng = wid % (G4 / 64);          // 0..71  (group of 4 n-tiles)
    int m0 = mt * 16, n0 = ng * 64;
    bool hiHalf = lane >= 16;
    int lm = lane & 15;

    v8f acc[4] = {v8f{}, v8f{}, v8f{}, v8f{}};
    const bf16_t* arow = A + (long)(m0 + lm) * DP;

    for (int kb = 0; kb < DP; kb += 32) {
        v16bf afrag = make_afrag(arow, kb, hiHalf);
        int bk = kb + (hiHalf ? 16 : 0);
#pragma unroll
        for (int t = 0; t < 4; ++t) {
            const bf16_t* brow = Bw + (long)(n0 + t * 16 + lm) * DP + bk;
            v16bf bfrag = *(const v16bf*)brow;
            acc[t] = __builtin_amdgcn_wmma_f32_16x16x32_bf16(
                false, afrag, false, bfrag, (short)0, acc[t], false, false);
        }
    }
#pragma unroll
    for (int t = 0; t < 4; ++t) {
        int n = n0 + t * 16 + lm;
        float bias = b_ih[n] + b_hh[n];
        int mr = m0 + (hiHalf ? 8 : 0);
#pragma unroll
        for (int r = 0; r < 8; ++r) {
            int mm = mr + r;                       // flat row = b*SS + s
            int b = mm >> 9, s = mm & (SS - 1);
            xg[((long)s * BB + b) * G4 + n] = (bf16_t)(acc[t][r] + bias);
        }
    }
}

// ---------------------------------------------------------------------------
// Phase 2: persistent recurrent kernel. 72 blocks x 128 threads.
// Block bx owns j-tile jt=bx (16 h-columns). Its 4 gate weight slabs
// (147456 B) are staged in LDS ONCE and reused for all 512 timesteps.
// The next timestep's xg tile (8 KB) is prefetched into an LDS double
// buffer by the Tensor Data Mover, overlapped with the WMMA K-loop.
// Wave w handles batch m-tile mt=w. Grid-wide step barrier via atomics.
// ---------------------------------------------------------------------------
__global__ void __launch_bounds__(128) lstm_steps(
    const bf16_t* __restrict__ xg,    // [SS, BB, G4] bf16
    const bf16_t* __restrict__ Whh,   // [G4, HH] bf16
    bf16_t* hbuf0, bf16_t* hbuf1,     // [BB, HH] bf16 ping-pong
    float* __restrict__ cbuf,         // [BB, HH] fp32, updated in place
    float* __restrict__ out,          // [BB, SS, HH] fp32
    int* sync_cnt)
{
    extern __shared__ char smem_raw[];
    bf16_t* lds = (bf16_t*)__builtin_assume_aligned(smem_raw, 64);

    const int tid = threadIdx.x;
    const int lane = tid & 31;
    const int mt = tid >> 5;           // wave id 0..3 -> batch tile
    const int jt = blockIdx.x;         // 0..71 -> h-column tile
    const bool hiHalf = lane >= 16;
    const int lm = lane & 15;
    const unsigned nblk = gridDim.x;

    // --- kick off TDM prefetch of the t=0 xg tile (wave 0 only; EXEC-free) ---
    if (mt == 0) {
        tdm_load_xg_tile(xg + (long)jt * 16, WSLAB_BYTES);
    }

    // --- stage this block's 64 weight rows (4 gates x 16 rows) into LDS ---
    {
        const int chunks = 64 * (HH / 8);          // v8bf chunks
        for (int idx = tid; idx < chunks; idx += blockDim.x) {
            int row = idx / (HH / 8);              // 0..63 : g*16 + r
            int kc  = (idx % (HH / 8)) * 8;
            int g = row >> 4, r = row & 15;
            long n = (long)g * HH + (long)jt * 16 + r;
            *(v8bf*)(lds + (long)row * HH + kc) = *(const v8bf*)(Whh + n * HH + kc);
        }
    }

    bf16_t* hprev = hbuf0;
    bf16_t* hnext = hbuf1;
    const int mb = mt * 16;

    for (int t = 0; t < SS; ++t) {
        // ensure this step's xg tile DMA has landed, then make it block-visible
        if (mt == 0) __builtin_amdgcn_s_wait_tensorcnt(0);
        __syncthreads();
        // prefetch next step's tile into the other LDS buffer
        if (mt == 0 && t + 1 < SS) {
            tdm_load_xg_tile(xg + ((long)(t + 1) * BB) * G4 + (long)jt * 16,
                             (unsigned)(WSLAB_BYTES + ((t + 1) & 1) * XGTILE_BYTES));
        }
        const bf16_t* xgt = (const bf16_t*)(smem_raw + WSLAB_BYTES + (t & 1) * XGTILE_BYTES);

        // init accumulators with gate preactivations (i,f,g,o order) from LDS
        v8f acc[4];
        {
            int mr = mb + (hiHalf ? 8 : 0);
#pragma unroll
            for (int g = 0; g < 4; ++g) {
#pragma unroll
                for (int r = 0; r < 8; ++r)
                    acc[g][r] = (float)xgt[(g * BB + mr + r) * 16 + lm];
            }
        }
        // K-loop: h_prev[64,1152] x Whh^T, B fragments from LDS weight slab
        const bf16_t* arow = hprev + (long)(mb + lm) * HH;
        for (int kb = 0; kb < HH; kb += 32) {
            v16bf afrag = make_afrag(arow, kb, hiHalf);
            int bk = kb + (hiHalf ? 16 : 0);
#pragma unroll
            for (int g = 0; g < 4; ++g) {
                v16bf bfrag = *(const v16bf*)(lds + (long)(g * 16 + lm) * HH + bk);
                acc[g] = __builtin_amdgcn_wmma_f32_16x16x32_bf16(
                    false, afrag, false, bfrag, (short)0, acc[g], false, false);
            }
        }
        // elementwise cell update
        {
            int j = jt * 16 + lm;
            int mr = mb + (hiHalf ? 8 : 0);
#pragma unroll
            for (int r = 0; r < 8; ++r) {
                int b = mr + r;
                float iv = sigf(acc[0][r]);
                float fv = sigf(acc[1][r]);
                float gv = tanhf(acc[2][r]);
                float ov = sigf(acc[3][r]);
                float cn = fv * cbuf[b * HH + j] + iv * gv;
                cbuf[b * HH + j] = cn;
                float hv = ov * tanhf(cn);
                out[((long)b * SS + t) * HH + j] = hv;
                hnext[b * HH + j] = (bf16_t)hv;
            }
        }
        // grid-wide step barrier (monotonic counter, no reset races)
        __syncthreads();
        if (tid == 0) {
            __threadfence();                        // release h/c writes to L2
            atomicAdd(sync_cnt, 1);
            int target = (int)((unsigned)(t + 1) * nblk);
            while (__hip_atomic_load(sync_cnt, __ATOMIC_ACQUIRE,
                                     __HIP_MEMORY_SCOPE_AGENT) < target) {}
        }
        __syncthreads();
        __threadfence();                            // acquire other blocks' h
        bf16_t* tmp = hprev; hprev = hnext; hnext = tmp;
    }
}

// ---------------------------------------------------------------------------
// Copy final (h, c) into the tail of d_out
// ---------------------------------------------------------------------------
__global__ void finalize(const float* __restrict__ out, const float* __restrict__ cbuf,
                         float* __restrict__ hdst, float* __restrict__ cdst) {
    int i = blockIdx.x * blockDim.x + threadIdx.x;
    if (i >= BB * HH) return;
    int b = i / HH, j = i % HH;
    hdst[i] = out[((long)b * SS + (SS - 1)) * HH + j];
    cdst[i] = cbuf[i];
}

// ---------------------------------------------------------------------------
extern "C" void kernel_launch(void* const* d_in, const int* in_sizes, int n_in,
                              void* d_out, int out_size, void* d_ws, size_t ws_size,
                              hipStream_t stream) {
    const float* inp  = (const float*)d_in[0];
    const float* h0   = (const float*)d_in[1];
    const float* c0   = (const float*)d_in[2];
    const float* w_ih = (const float*)d_in[3];
    const float* w_hh = (const float*)d_in[4];
    const float* b_ih = (const float*)d_in[5];
    const float* b_hh = (const float*)d_in[6];
    float* out = (float*)d_out;

    // workspace carve-out (256B aligned)
    char* ws = (char*)d_ws;
    size_t off = 0;
    auto carve = [&](size_t bytes) -> void* {
        void* p = ws + off;
        off = (off + bytes + 255) & ~(size_t)255;
        return p;
    };
    bf16_t* xg    = (bf16_t*)carve((size_t)MM * G4 * 2);   // 302 MB, [S,B,4H]
    bf16_t* inpb  = (bf16_t*)carve((size_t)MM * DP * 2);   //  27 MB
    bf16_t* wihb  = (bf16_t*)carve((size_t)G4 * DP * 2);   // 3.8 MB
    bf16_t* whhb  = (bf16_t*)carve((size_t)G4 * HH * 2);   // 10.6 MB
    bf16_t* hbuf0 = (bf16_t*)carve((size_t)BB * HH * 2);
    bf16_t* hbuf1 = (bf16_t*)carve((size_t)BB * HH * 2);
    float*  cbuf  = (float*) carve((size_t)BB * HH * 4);
    int*    sync  = (int*)   carve(256);

    // 1. stage bf16 (padded) copies of inp, w_ih, w_hh
    {
        long n = (long)MM * DP;
        cvt_pad_bf16<<<(unsigned)((n + 255) / 256), 256, 0, stream>>>(inp, inpb, MM, DD, DP);
    }
    {
        long n = (long)G4 * DP;
        cvt_pad_bf16<<<(unsigned)((n + 255) / 256), 256, 0, stream>>>(w_ih, wihb, G4, DD, DP);
    }
    {
        long n = (long)G4 * HH;
        cvt_pad_bf16<<<(unsigned)((n + 255) / 256), 256, 0, stream>>>(w_hh, whhb, G4, HH, HH);
    }
    // 2. init h/c state + sync counter
    init_state<<<(BB * HH + 255) / 256, 256, 0, stream>>>(h0, c0, hbuf0, cbuf, sync);

    // 3. big input-side GEMM: 2048 m-tiles x 72 n-groups, 8 waves/block
    {
        int waves = (MM / 16) * (G4 / 64);          // 147456
        gemm_xg<<<waves / 8, 256, 0, stream>>>(inpb, wihb, b_ih, b_hh, xg);
    }

    // 4. persistent recurrent scan: 72 blocks, 163840 B dynamic LDS each
    {
        size_t ldsBytes = (size_t)LDS_TOTAL;
        hipFuncSetAttribute((const void*)lstm_steps,
                            hipFuncAttributeMaxDynamicSharedMemorySize,
                            (int)ldsBytes);
        lstm_steps<<<HH / 16, 128, ldsBytes, stream>>>(
            xg, whhb, hbuf0, hbuf1, cbuf, out, sync);
    }

    // 5. final (h, c) into d_out tail
    {
        float* hdst = out + (size_t)BB * SS * HH;
        float* cdst = hdst + (size_t)BB * HH;
        finalize<<<(BB * HH + 255) / 256, 256, 0, stream>>>(out, cbuf, hdst, cdst);
    }
}
